// DFSHA_77618648973711
// MI455X (gfx1250) — 16-bit and 8-bit matrix operand order repaired and checked by reading, compiled, not run on hardware
//
#include <hip/hip_runtime.h>
#include <hip/hip_bf16.h>


typedef float  v8f  __attribute__((ext_vector_type(8)));
typedef __bf16 v16bf __attribute__((ext_vector_type(16)));
typedef int    v8i  __attribute__((ext_vector_type(8)));

__device__ __forceinline__ __bf16 f2bf(float f) { return (__bf16)f; }

__global__ __launch_bounds__(128) void gemm_bf16_k(
    const float* __restrict__ A, long astr,
    const float* __restrict__ Bm, long bstr, int ldb, int transB,
    float* __restrict__ C, long cstr,
    const float* __restrict__ bias,
    const float* __restrict__ resid, long rstr,
    int M, int N, int K)
{
  const int lane = threadIdx.x;
  const int z    = blockIdx.z;
  const int tile_m = blockIdx.y * 64 + threadIdx.y * 16;
  const int tile_n = blockIdx.x * 16;
  const float* Ab = A  + astr * (long)z;
  const float* Bb = Bm + bstr * (long)z;
  const int l15 = lane & 15;
  const int kh  = lane >> 4;
  const int arow = tile_m + l15;
  const int bcol = tile_n + l15;
  v8f acc = {};
  for (int k0 = 0; k0 < K; k0 += 32) {
    v16bf af, bfr;
    const float* ap = Ab + (long)arow * K + k0 + kh * 16;
#pragma unroll
    for (int j = 0; j < 4; ++j) {
      af[2*j]     = f2bf(ap[2*j]);
      af[2*j+1]   = f2bf(ap[2*j+1]);
      af[8+2*j]   = f2bf(ap[8+2*j]);
      af[8+2*j+1] = f2bf(ap[8+2*j+1]);
    }
    if (transB) {
      const float* bp = Bb + (long)bcol * ldb + k0 + kh * 16;
#pragma unroll
      for (int v = 0; v < 8; ++v) { bfr[2*v] = f2bf(bp[2*v]); bfr[2*v+1] = f2bf(bp[2*v+1]); }
    } else {
      const float* bp = Bb + (long)(k0 + kh * 16) * ldb + bcol;
#pragma unroll
      for (int v = 0; v < 8; ++v) {
        bfr[2*v]   = f2bf(bp[(long)(2*v)   * ldb]);
        bfr[2*v+1] = f2bf(bp[(long)(2*v+1) * ldb]);
      }
    }
    acc = __builtin_amdgcn_wmma_f32_16x16x32_bf16(false, af, false, bfr,
                                                  (short)0, acc, false, false);
  }
#pragma unroll
  for (int j = 0; j < 8; ++j) {
    int row = tile_m + j + kh * 8;
    float v = acc[j];
    if (bias)  v += bias[row];
    long off = (long)row * N + (tile_n + l15);
    if (resid) v += resid[rstr * (long)z + off];
    C[cstr * (long)z + off] = v;
  }
}

__global__ __launch_bounds__(128) void gemm_i8_attn_k(
    const signed char* __restrict__ QBT,
    const signed char* __restrict__ KB,
    float* __restrict__ A2, float scale)
{
  const int lane = threadIdx.x;
  const int z = blockIdx.z;
  const int tile_m = blockIdx.y * 64 + threadIdx.y * 16;
  const int tile_n = blockIdx.x * 16;
  const signed char* Qb = QBT + (long)z * 1024 * 128;
  const signed char* Kb = KB  + (long)z * 128 * 1024;
  const int l15 = lane & 15, kh = lane >> 4;
  const int arow = tile_m + l15;
  const int bcol = tile_n + l15;
  v8i acc = {};
#pragma unroll
  for (int k0 = 0; k0 < 128; k0 += 64) {
    v8i af, bfr;
#pragma unroll
    for (int v = 0; v < 8; ++v) {
      int k = k0 + (v >> 1) * 16 + kh * 8 + (v & 1) * 4;
      int w; __builtin_memcpy(&w, Qb + (long)arow * 128 + k, 4);
      af[v] = w;
    }
#pragma unroll
    for (int v = 0; v < 8; ++v) {
      int k = k0 + (v >> 1) * 16 + kh * 8 + (v & 1) * 4;
      const signed char* p = Kb + (long)k * 1024 + bcol;
      unsigned b0 = (unsigned char)p[0],    b1 = (unsigned char)p[1024];
      unsigned b2 = (unsigned char)p[2048], b3 = (unsigned char)p[3072];
      bfr[v] = (int)(b0 | (b1 << 8) | (b2 << 16) | (b3 << 24));
    }
    acc = __builtin_amdgcn_wmma_i32_16x16x64_iu8(true, af, true, bfr,
                                                 acc, false, false);
  }
#pragma unroll
  for (int j = 0; j < 8; ++j) {
    int row = tile_m + j + kh * 8;
    A2[(long)z * 1048576 + (long)row * 1024 + tile_n + l15] = (float)acc[j] * scale;
  }
}

__global__ __launch_bounds__(256) void fft_pool_k(const float* __restrict__ YC,
                                                  float* __restrict__ pooled)
{
  const int bc = blockIdx.x, b = bc >> 7, c = bc & 127;
  const float* img = YC + (long)b * 393216 + (long)(128 + c) * 1024;
  __shared__ float yt[1024], gre[1024], gim[1024];
  __shared__ float ct[32], st[32];
  __shared__ float red[256];
  int t = threadIdx.x;
  for (int i = t; i < 1024; i += 256) yt[i] = img[i];
  if (t < 32) { float a = 6.28318530717958647f * (float)t / 32.f; ct[t] = cosf(a); st[t] = sinf(a); }
  __syncthreads();
  for (int i = t; i < 1024; i += 256) {
    int k = i >> 5, n = i & 31;
    float re = 0.f, im = 0.f;
    for (int m = 0; m < 32; ++m) {
      int ph = (k * m) & 31; float v = yt[m * 32 + n];
      re += v * ct[ph]; im -= v * st[ph];
    }
    gre[k * 32 + n] = re; gim[k * 32 + n] = im;
  }
  __syncthreads();
  float s = 0.f;
  for (int i = t; i < 544; i += 256) {
    int k = i / 17, l = i % 17;
    float xr = 0.f, xi = 0.f;
    for (int n = 0; n < 32; ++n) {
      int ph = (l * n) & 31;
      float a = gre[k * 32 + n], bb = gim[k * 32 + n];
      xr += a * ct[ph] + bb * st[ph];
      xi += bb * ct[ph] - a * st[ph];
    }
    s += sqrtf(xr * xr + xi * xi);
  }
  red[t] = s; __syncthreads();
  for (int o = 128; o > 0; o >>= 1) { if (t < o) red[t] += red[t + o]; __syncthreads(); }
  if (t == 0) pooled[bc] = red[0] * (1.0f / 32.0f) / 544.0f;
}

__global__ __launch_bounds__(256) void colmean_k(const float* __restrict__ YC,
                                                 float* __restrict__ ymean)
{
  const int bc = blockIdx.x, b = bc >> 7, c = bc & 127;
  const float* row = YC + (long)b * 393216 + (long)(128 + c) * 1024;
  __shared__ float red[256];
  int t = threadIdx.x; float s = 0.f;
  for (int i = t; i < 1024; i += 256) s += row[i];
  red[t] = s; __syncthreads();
  for (int o = 128; o > 0; o >>= 1) { if (t < o) red[t] += red[t + o]; __syncthreads(); }
  if (t == 0) ymean[bc] = red[0] * (1.0f / 1024.0f);
}

__global__ __launch_bounds__(128) void fwca_k(
    const float* __restrict__ pooled, const float* __restrict__ ymean,
    const float* __restrict__ fm_w1, const float* __restrict__ fm_w2,
    const float* __restrict__ ca_w1, const float* __restrict__ ca_w2,
    const float* __restrict__ rt_w,  const float* __restrict__ rt_b,
    float* __restrict__ G, float* __restrict__ ALPHA)
{
  const int b = blockIdx.x, t = threadIdx.x;
  __shared__ float pl[128], h1[8], xm[128], h2[8];
  pl[t] = pooled[b * 128 + t];
  __syncthreads();
  if (t < 8) { float s = 0.f; for (int c = 0; c < 128; ++c) s += pl[c] * fm_w1[t * 128 + c]; h1[t] = fmaxf(s, 0.f); }
  __syncthreads();
  float s = 0.f;
  for (int r = 0; r < 8; ++r) s += h1[r] * fm_w2[t * 8 + r];
  float fwv = 1.f / (1.f + __expf(-s));
  xm[t] = fwv * ymean[b * 128 + t];
  __syncthreads();
  if (t < 8) { float s2 = 0.f; for (int c = 0; c < 128; ++c) s2 += xm[c] * ca_w1[t * 128 + c]; h2[t] = fmaxf(s2, 0.f); }
  __syncthreads();
  float s3 = 0.f;
  for (int r = 0; r < 8; ++r) s3 += h2[r] * ca_w2[t * 8 + r];
  float cav = 1.f / (1.f + __expf(-s3));
  G[b * 128 + t] = fwv * cav;
  if (t == 0) {
    float pr = 0.f;
    for (int c = 0; c < 128; ++c) pr += ymean[b * 128 + c] * rt_w[c];
    ALPHA[b] = 1.f / (1.f + __expf(-(pr + rt_b[0])));
  }
}

__global__ __launch_bounds__(256) void var_k(const float* __restrict__ QKV,
                                             float* __restrict__ varv,
                                             float* __restrict__ vark)
{
  int idx = blockIdx.x * 256 + threadIdx.x;
  if (idx >= 65536) return;
  int n = idx & 1023, h = (idx >> 10) & 7, b = idx >> 13;
  const float* base = QKV + (long)b * 393216;
  {
    const float* p = base + (long)(256 + h * 16) * 1024 + n;
    float s = 0.f, ss = 0.f;
    for (int d = 0; d < 16; ++d) { float x = p[d * 1024]; s += x; ss += x * x; }
    float mu = s * (1.f / 16.f);
    varv[idx] = (ss - 16.f * mu * mu) * (1.f / 15.f);
  }
  {
    const float* p = base + (long)(128 + h * 16) * 1024 + n;
    float s = 0.f, ss = 0.f;
    for (int d = 0; d < 16; ++d) { float x = p[d * 1024]; s += x; ss += x * x; }
    float mu = s * (1.f / 16.f);
    vark[idx] = (ss - 16.f * mu * mu) * (1.f / 15.f);
  }
}

__global__ __launch_bounds__(256) void minmax_k(const float* __restrict__ vark,
                                                float* __restrict__ tmin,
                                                float* __restrict__ tmax)
{
  const int r = blockIdx.x;
  const float* p = vark + (long)r * 1024;
  __shared__ float rmn[256], rmx[256];
  int t = threadIdx.x;
  float mn = 3.4e38f, mx = -3.4e38f;
  for (int i = t; i < 1024; i += 256) { float v = p[i]; mn = fminf(mn, v); mx = fmaxf(mx, v); }
  rmn[t] = mn; rmx[t] = mx; __syncthreads();
  for (int o = 128; o > 0; o >>= 1) {
    if (t < o) { rmn[t] = fminf(rmn[t], rmn[t + o]); rmx[t] = fmaxf(rmx[t], rmx[t + o]); }
    __syncthreads();
  }
  if (t == 0) { tmin[r] = rmn[0]; tmax[r] = rmx[0]; }
}

__global__ __launch_bounds__(128) void statattn_k(
    const float* __restrict__ QKV,
    const float* __restrict__ varv, const float* __restrict__ vark,
    const float* __restrict__ tmin, const float* __restrict__ tmax,
    float* __restrict__ TO)
{
  int wid = blockIdx.x * 4 + threadIdx.y;
  int lane = threadIdx.x;
  int n = wid & 1023, h = (wid >> 10) & 7, b = wid >> 13;
  int bh = b * 8 + h;
  float s = varv[wid] * 0.25f;
  float mrow = (s >= 0.f) ? s * tmax[bh] : s * tmin[bh];
  const float* tv = vark + (long)bh * 1024;
  const float* vbase = QKV + (long)b * 393216 + (long)(256 + h * 16) * 1024;
  float acc[16];
#pragma unroll
  for (int d = 0; d < 16; ++d) acc[d] = 0.f;
  float den = 0.f;
  for (int m = lane; m < 1024; m += 32) {
    float e = __expf(s * tv[m] - mrow);
    den += e;
#pragma unroll
    for (int d = 0; d < 16; ++d) acc[d] += e * vbase[d * 1024 + m];
  }
#pragma unroll
  for (int o = 16; o > 0; o >>= 1) {
    den += __shfl_xor(den, o, 32);
#pragma unroll
    for (int d = 0; d < 16; ++d) acc[d] += __shfl_xor(acc[d], o, 32);
  }
  if (lane == 0) {
    float inv = 1.f / den;
    float* outp = TO + (long)b * 131072 + (long)(h * 16) * 1024 + n;
#pragma unroll
    for (int d = 0; d < 16; ++d) outp[d * 1024] = acc[d] * inv;
  }
}

__global__ __launch_bounds__(256) void binarize_k(
    const float* __restrict__ QP, const float* __restrict__ KP,
    signed char* __restrict__ QBT, signed char* __restrict__ KB)
{
  long i = (long)blockIdx.x * 256 + threadIdx.x;
  if (i >= 1048576) return;
  int n = (int)(i & 1023), c = (int)((i >> 10) & 127); long b = i >> 17;
  float q = QP[i], k = KP[i];
  QBT[b * 131072 + (long)n * 128 + c] = (q > 0.f) ? 1 : ((q < 0.f) ? -1 : 0);
  KB[i]                               = (k > 0.f) ? 1 : ((k < 0.f) ? -1 : 0);
}

__global__ __launch_bounds__(256) void softmax_k(float* __restrict__ A2)
{
  float* row = A2 + (long)blockIdx.x * 1024;
  __shared__ float red[256];
  int t = threadIdx.x;
  float v0 = row[t], v1 = row[t + 256], v2 = row[t + 512], v3 = row[t + 768];
  float mx = fmaxf(fmaxf(v0, v1), fmaxf(v2, v3));
  red[t] = mx; __syncthreads();
  for (int o = 128; o > 0; o >>= 1) { if (t < o) red[t] = fmaxf(red[t], red[t + o]); __syncthreads(); }
  mx = red[0]; __syncthreads();
  v0 = __expf(v0 - mx); v1 = __expf(v1 - mx); v2 = __expf(v2 - mx); v3 = __expf(v3 - mx);
  red[t] = v0 + v1 + v2 + v3; __syncthreads();
  for (int o = 128; o > 0; o >>= 1) { if (t < o) red[t] += red[t + o]; __syncthreads(); }
  float inv = 1.f / red[0];
  row[t] = v0 * inv; row[t + 256] = v1 * inv; row[t + 512] = v2 * inv; row[t + 768] = v3 * inv;
}

__global__ __launch_bounds__(256) void combine_k(
    float* __restrict__ YC, const float* __restrict__ STAT,
    const float* __restrict__ OUTBIN,
    const float* __restrict__ G, const float* __restrict__ ALPHA,
    const float* __restrict__ dw_w, const float* __restrict__ dw_b)
{
  const int bc = blockIdx.x, b = bc >> 7, c = bc & 127;
  const float* y1  = YC + (long)b * 393216 + (long)(128 + c) * 1024;
  float*       outp = YC + (long)b * 393216 + (long)(256 + c) * 1024;
  const float* st = STAT   + (long)bc * 1024;
  const float* ob = OUTBIN + (long)bc * 1024;
  float g = G[bc], a = ALPHA[b], db = dw_b[c];
  float w[9];
#pragma unroll
  for (int i = 0; i < 9; ++i) w[i] = dw_w[c * 9 + i];
  for (int n = threadIdx.x; n < 1024; n += 256) {
    int hh = n >> 5, ww = n & 31;
    float conv = 0.f;
#pragma unroll
    for (int dy = -1; dy <= 1; ++dy)
#pragma unroll
      for (int dx = -1; dx <= 1; ++dx) {
        int yy = hh + dy, xx = ww + dx;
        if (yy >= 0 && yy < 32 && xx >= 0 && xx < 32)
          conv += y1[yy * 32 + xx] * w[(dy + 1) * 3 + dx + 1];
      }
    outp[n] = g * y1[n] + st[n] + (1.f - a) * (conv + db) + a * ob[n];
  }
}

extern "C" void kernel_launch(void* const* d_in, const int* in_sizes, int n_in,
                              void* d_out, int out_size, void* d_ws, size_t ws_size,
                              hipStream_t stream) {
  (void)in_sizes; (void)n_in; (void)out_size; (void)ws_size;
  const float* x       = (const float*)d_in[0];
  const float* cv1_w   = (const float*)d_in[1];
  const float* cv1_b   = (const float*)d_in[2];
  const float* fm_w1   = (const float*)d_in[3];
  const float* fm_w2   = (const float*)d_in[4];
  const float* ca_w1   = (const float*)d_in[5];
  const float* ca_w2   = (const float*)d_in[6];
  const float* qkv_w   = (const float*)d_in[7];
  const float* tproj_w = (const float*)d_in[8];
  const float* tproj_b = (const float*)d_in[9];
  const float* dw_w    = (const float*)d_in[10];
  const float* dw_b    = (const float*)d_in[11];
  const float* rt_w    = (const float*)d_in[12];
  const float* rt_b    = (const float*)d_in[13];
  const float* q_w     = (const float*)d_in[14];
  const float* k_w     = (const float*)d_in[15];
  const float* v_w     = (const float*)d_in[16];
  const float* sproj_w = (const float*)d_in[17];
  const float* sproj_b = (const float*)d_in[18];
  const float* cv2_w   = (const float*)d_in[19];
  const float* cv2_b   = (const float*)d_in[20];
  float* out = (float*)d_out;
  char* ws = (char*)d_ws;

  float* YC   = (float*)(ws + 0);
  float* QKV  = (float*)(ws + 12582912);
  float* TO   = (float*)(ws + 25165824);
  float* STAT = (float*)(ws + 29360128);
  float* QP   = (float*)(ws + 33554432);
  float* KP   = (float*)(ws + 37748736);
  float* VP   = (float*)(ws + 41943040);
  float* A2   = (float*)(ws + 46137344);
  signed char* QBT = (signed char*)(ws + 79691776);
  signed char* KB  = (signed char*)(ws + 80740352);
  float* VARV = (float*)(ws + 81788928);
  float* VARK = (float*)(ws + 82051072);
  float* TMIN = (float*)(ws + 82313216);
  float* TMAX = (float*)(ws + 82313472);
  float* POOL = (float*)(ws + 82313728);
  float* YMEAN= (float*)(ws + 82317824);
  float* Gg   = (float*)(ws + 82321920);
  float* ALPHA= (float*)(ws + 82326016);

  const dim3 wblk(32, 4, 1);

  gemm_bf16_k<<<dim3(64, 4, 8), wblk, 0, stream>>>(cv1_w, 0, x, 262144, 1024, 0,
      YC, 393216, cv1_b, nullptr, 0, 256, 1024, 256);

  fft_pool_k<<<dim3(1024), dim3(256), 0, stream>>>(YC, POOL);
  colmean_k<<<dim3(1024), dim3(256), 0, stream>>>(YC, YMEAN);
  fwca_k<<<dim3(8), dim3(128), 0, stream>>>(POOL, YMEAN, fm_w1, fm_w2, ca_w1, ca_w2,
                                            rt_w, rt_b, Gg, ALPHA);

  gemm_bf16_k<<<dim3(64, 6, 8), wblk, 0, stream>>>(qkv_w, 0, YC + 131072, 393216, 1024, 0,
      QKV, 393216, nullptr, nullptr, 0, 384, 1024, 128);

  var_k<<<dim3(256), dim3(256), 0, stream>>>(QKV, VARV, VARK);
  minmax_k<<<dim3(64), dim3(256), 0, stream>>>(VARK, TMIN, TMAX);
  statattn_k<<<dim3(16384), dim3(32, 4), 0, stream>>>(QKV, VARV, VARK, TMIN, TMAX, TO);
  gemm_bf16_k<<<dim3(64, 2, 8), wblk, 0, stream>>>(tproj_w, 0, TO, 131072, 1024, 0,
      STAT, 131072, tproj_b, nullptr, 0, 128, 1024, 128);

  gemm_bf16_k<<<dim3(64, 2, 8), wblk, 0, stream>>>(q_w, 0, YC + 131072, 393216, 1024, 0,
      QP, 131072, nullptr, nullptr, 0, 128, 1024, 128);
  gemm_bf16_k<<<dim3(64, 2, 8), wblk, 0, stream>>>(k_w, 0, YC + 131072, 393216, 1024, 0,
      KP, 131072, nullptr, nullptr, 0, 128, 1024, 128);
  gemm_bf16_k<<<dim3(64, 2, 8), wblk, 0, stream>>>(v_w, 0, YC + 131072, 393216, 1024, 0,
      VP, 131072, nullptr, nullptr, 0, 128, 1024, 128);
  binarize_k<<<dim3(4096), dim3(256), 0, stream>>>(QP, KP, QBT, KB);

  gemm_i8_attn_k<<<dim3(64, 16, 8), wblk, 0, stream>>>(QBT, KB, A2, 0.0883883476483184f);
  softmax_k<<<dim3(8192), dim3(256), 0, stream>>>(A2);

  float* OB = QP; float* OUTBIN = KP;
  gemm_bf16_k<<<dim3(64, 2, 8), wblk, 0, stream>>>(VP, 131072, A2, 1048576, 1024, 1,
      OB, 131072, nullptr, nullptr, 0, 128, 1024, 1024);
  gemm_bf16_k<<<dim3(64, 2, 8), wblk, 0, stream>>>(sproj_w, 0, OB, 131072, 1024, 0,
      OUTBIN, 131072, sproj_b, nullptr, 0, 128, 1024, 128);

  combine_k<<<dim3(1024), dim3(256), 0, stream>>>(YC, STAT, OUTBIN, Gg, ALPHA, dw_w, dw_b);

  gemm_bf16_k<<<dim3(64, 4, 8), wblk, 0, stream>>>(cv2_w, 0, YC, 393216, 1024, 0,
      out, 262144, cv2_b, x, 262144, 256, 1024, 384);
}
